// SceneGuided_ConGNN_1288490189685
// MI455X (gfx1250) — compile-verified, matrix operand order checked by reading, not run on hardware
//
#include <hip/hip_runtime.h>

// ---------------- problem constants ----------------
#define DD   512
#define HH   4
#define CC   128
#define BN   256
#define FN   48
#define ON   64
#define NF   12288          // BN*FN
#define NO   16384          // BN*ON
#define NKF  73728          // NF*6
#define EF   86016          // NKF + NF self loops
#define NKO  98304          // NO*6
#define EO   114688         // NKO + NO
#define LSEQ 112            // FN + ON

// ---------------- helpers ----------------
__device__ __forceinline__ unsigned f2u(float f){ union{float f;unsigned u;}x; x.f=f; return x.u; }
__device__ __forceinline__ float u2f(unsigned u){ union{float f;unsigned u;}x; x.u=u; return x.f; }
__device__ __forceinline__ unsigned short f2bf(float f){
  unsigned u = f2u(f);
  u += 0x7fffu + ((u >> 16) & 1u);
  return (unsigned short)(u >> 16);
}
__device__ __forceinline__ unsigned fenc(float f){
  unsigned u = f2u(f);
  return (u & 0x80000000u) ? ~u : (u | 0x80000000u);
}
__device__ __forceinline__ float fdec(unsigned u){
  return u2f((u & 0x80000000u) ? (u & 0x7fffffffu) : ~u);
}

typedef __attribute__((ext_vector_type(16))) __bf16 v16bf;
typedef __attribute__((ext_vector_type(8)))  float  v8f;
union FragAB { v16bf v; unsigned u[8]; };
union FragC  { v8f  v; float   f[8]; };

// ---------------- WMMA GEMM: C[M,N] = act(A_bf16[M,K] @ B_bf16[K,N] + bias) ----------------
// M%128==0, N%128==0, K%32==0. act: 0=none 1=relu 2=sigmoid
// A tile staged via global_load_async_to_lds_b128 (ASYNCcnt path); B tile transposed via VALU.
__global__ __launch_bounds__(256) void wmma_gemm(
    const unsigned short* __restrict__ A, const unsigned short* __restrict__ B,
    const float* __restrict__ bias, float* __restrict__ C,
    int M, int N, int K, int act)
{
  __shared__ unsigned lds_a[2048];   // [m 0..127][kpair 0..15]
  __shared__ unsigned lds_b[2048];   // [n 0..127][kpair 0..15] (B transposed)
  const int tid  = threadIdx.x;
  const int lane = tid & 31;
  const int wave = tid >> 5;
  const int wm   = wave & 3;           // 4 waves along M
  const int wn   = wave >> 2;          // 2 waves along N
  const int half = lane >> 4;
  const int l15  = lane & 15;
  const int m0   = blockIdx.y * 128;
  const int n0   = blockIdx.x * 128;
  const unsigned* Ag = (const unsigned*)A;
  const unsigned* Bg = (const unsigned*)B;
  const int Kp = K >> 1, Np = N >> 1;
  const unsigned lds_a_base = (unsigned)(size_t)(&lds_a[0]);  // LDS byte offset = addr[31:0]

  FragC acc[2][4];
  #pragma unroll
  for (int i=0;i<2;i++)
    #pragma unroll
    for (int j=0;j<4;j++)
      #pragma unroll
      for (int e=0;e<8;e++) acc[i][j].f[e] = 0.f;

  for (int k0 = 0; k0 < K; k0 += 32) {
    // ---- stage A tile (128 x 32 bf16) via async copy: 512 x 16B chunks ----
    #pragma unroll
    for (int i = tid; i < 512; i += 256) {
      int m = i >> 2, c = i & 3;                      // 4 x 16B chunks per row
      const unsigned* g = Ag + (size_t)(m0 + m) * Kp + (k0 >> 1) + c * 4;
      unsigned loff = lds_a_base + (unsigned)(m * 64 + c * 16);
      asm volatile("global_load_async_to_lds_b128 %0, %1, off"
                   :: "v"(loff), "v"(g)
                   : "memory");
    }
    // ---- stage B tile (32 x 128 bf16), transposed into [n][k] ----
    #pragma unroll
    for (int i = tid; i < 2048; i += 256) {
      int k = i >> 6, np = i & 63;
      unsigned w2 = Bg[(size_t)(k0 + k) * Np + (n0 >> 1) + np];
      unsigned short* lb = (unsigned short*)lds_b;
      lb[(2*np)   * 32 + k] = (unsigned short)(w2 & 0xffffu);
      lb[(2*np+1) * 32 + k] = (unsigned short)(w2 >> 16);
    }
    if (k0 + 32 < K) {  // hint next B slab (global_prefetch_b8)
      __builtin_prefetch(Bg + (size_t)(k0 + 32 + (tid >> 3)) * Np + (n0 >> 1), 0, 0);
    }
    asm volatile("s_wait_asynccnt 0x0" ::: "memory");  // A-tile async copies complete
    __syncthreads();

    FragAB a[2], b[4];
    #pragma unroll
    for (int i=0;i<2;i++){
      int m = wm*32 + i*16 + l15;
      #pragma unroll
      for (int j=0;j<8;j++){
        int kidx = (j<4) ? (half*4 + j) : (8 + half*4 + (j-4));
        a[i].u[j] = lds_a[m*16 + kidx];
      }
    }
    #pragma unroll
    for (int j=0;j<4;j++){
      int n = wn*64 + j*16 + l15;
      #pragma unroll
      for (int jj=0;jj<8;jj++){
        int kidx = (jj<4) ? (half*4 + jj) : (8 + half*4 + (jj-4));
        b[j].u[jj] = lds_b[n*16 + kidx];
      }
    }
    #pragma unroll
    for (int i=0;i<2;i++)
      #pragma unroll
      for (int j=0;j<4;j++)
        acc[i][j].v = __builtin_amdgcn_wmma_f32_16x16x32_bf16(
            false, a[i].v, false, b[j].v, (short)0, acc[i][j].v, false, false);
    __syncthreads();
  }

  #pragma unroll
  for (int i=0;i<2;i++)
    #pragma unroll
    for (int j=0;j<4;j++){
      int col = n0 + wn*64 + j*16 + l15;
      float bv = bias ? bias[col] : 0.f;
      #pragma unroll
      for (int e=0;e<8;e++){
        int row = m0 + wm*32 + i*16 + half*8 + e;
        float v = acc[i][j].f[e] + bv;
        if (act == 1) v = fmaxf(v, 0.f);
        else if (act == 2) v = 1.f/(1.f + __expf(-v));
        C[(size_t)row * N + col] = v;
      }
    }
}

// ---------------- LN over rows -> bf16 (pre-GEMM LN of _red inputs) ----------------
__global__ __launch_bounds__(256) void ln_rows_bf16(
    const float* __restrict__ X, const float* __restrict__ g, const float* __restrict__ b,
    unsigned short* __restrict__ Y, int W)
{
  int row = blockIdx.x, tid = threadIdx.x;
  const float* x = X + (size_t)row * W;
  __shared__ float r1[256], r2[256];
  float s=0.f, s2=0.f;
  for (int i=tid;i<W;i+=256){ float v=x[i]; s+=v; s2+=v*v; }
  r1[tid]=s; r2[tid]=s2; __syncthreads();
  for (int o=128;o;o>>=1){ if (tid<o){ r1[tid]+=r1[tid+o]; r2[tid]+=r2[tid+o]; } __syncthreads(); }
  float inv = 1.f/(float)W;
  float mean = r1[0]*inv;
  float rstd = rsqrtf(r2[0]*inv - mean*mean + 1e-5f);
  for (int i=tid;i<W;i+=256)
    Y[(size_t)row*W + i] = f2bf((x[i]-mean)*rstd*g[i] + b[i]);
}

// ---------------- LN(width 512, affine) + optional relu -> f32 (+optional bf16) ----------------
__global__ __launch_bounds__(256) void ln_act_dual(
    const float* __restrict__ X, const float* __restrict__ g, const float* __restrict__ b,
    float* __restrict__ Y, unsigned short* __restrict__ Ybf, int act)
{
  int row = blockIdx.x, tid = threadIdx.x;
  const float* x = X + (size_t)row * DD;
  __shared__ float r1[256], r2[256];
  float s=0.f, s2=0.f;
  for (int i=tid;i<DD;i+=256){ float v=x[i]; s+=v; s2+=v*v; }
  r1[tid]=s; r2[tid]=s2; __syncthreads();
  for (int o=128;o;o>>=1){ if (tid<o){ r1[tid]+=r1[tid+o]; r2[tid]+=r2[tid+o]; } __syncthreads(); }
  float mean = r1[0]*(1.f/DD);
  float rstd = rsqrtf(r2[0]*(1.f/DD) - mean*mean + 1e-5f);
  for (int i=tid;i<DD;i+=256){
    float v = (x[i]-mean)*rstd*g[i] + b[i];
    if (act == 1) v = fmaxf(v, 0.f);
    Y[(size_t)row*DD + i] = v;
    if (Ybf) Ybf[(size_t)row*DD + i] = f2bf(v);
  }
}

// ---------------- fs = LN_plain(x1 + x2), width 512 ----------------
__global__ __launch_bounds__(256) void add_lnplain(
    const float* __restrict__ X1, const float* __restrict__ X2, float* __restrict__ Y)
{
  int row = blockIdx.x, tid = threadIdx.x;
  const float* a = X1 + (size_t)row * DD;
  const float* c = X2 + (size_t)row * DD;
  __shared__ float r1[256], r2[256];
  float s=0.f, s2=0.f;
  for (int i=tid;i<DD;i+=256){ float v=a[i]+c[i]; s+=v; s2+=v*v; }
  r1[tid]=s; r2[tid]=s2; __syncthreads();
  for (int o=128;o;o>>=1){ if (tid<o){ r1[tid]+=r1[tid+o]; r2[tid]+=r2[tid+o]; } __syncthreads(); }
  float mean = r1[0]*(1.f/DD);
  float rstd = rsqrtf(r2[0]*(1.f/DD) - mean*mean + 1e-5f);
  for (int i=tid;i<DD;i+=256)
    Y[(size_t)row*DD + i] = (a[i]+c[i]-mean)*rstd;
}

// ---------------- GAT post: h_out = h_in + elu(LN(gat+bias, ng, nb)) ----------------
__global__ __launch_bounds__(256) void gat_post(
    const float* __restrict__ h_in, const float* __restrict__ gat, const float* __restrict__ bias,
    const float* __restrict__ ng, const float* __restrict__ nb,
    float* __restrict__ h_out, unsigned short* __restrict__ h_out_bf)
{
  int row = blockIdx.x, tid = threadIdx.x;
  const float* gr = gat + (size_t)row * DD;
  __shared__ float r1[256], r2[256];
  float s=0.f, s2=0.f;
  for (int i=tid;i<DD;i+=256){ float v=gr[i]+bias[i]; s+=v; s2+=v*v; }
  r1[tid]=s; r2[tid]=s2; __syncthreads();
  for (int o=128;o;o>>=1){ if (tid<o){ r1[tid]+=r1[tid+o]; r2[tid]+=r2[tid+o]; } __syncthreads(); }
  float mean = r1[0]*(1.f/DD);
  float rstd = rsqrtf(r2[0]*(1.f/DD) - mean*mean + 1e-5f);
  for (int i=tid;i<DD;i+=256){
    float v = (gr[i]+bias[i]-mean)*rstd*ng[i] + nb[i];
    v = v > 0.f ? v : expm1f(v);
    float o = h_in[(size_t)row*DD + i] + v;
    h_out[(size_t)row*DD + i] = o;
    h_out_bf[(size_t)row*DD + i] = f2bf(o);
  }
}

// ---------------- hf2 = LN_plain(hf + gate * (alpha * fs[batch])) ----------------
__global__ __launch_bounds__(256) void face_fuse(
    const float* __restrict__ hf, const float* __restrict__ gate, const float* __restrict__ alpha,
    const float* __restrict__ fs, const int* __restrict__ fbatch, float* __restrict__ out)
{
  int row = blockIdx.x, tid = threadIdx.x;
  int bg = fbatch[row];
  const float* h = hf + (size_t)row * DD;
  const float* gt = gate + (size_t)row * DD;
  const float* f = fs + (size_t)bg * DD;
  __shared__ float r1[256], r2[256];
  float s=0.f, s2=0.f;
  for (int i=tid;i<DD;i+=256){ float v=h[i]+gt[i]*(alpha[i]*f[i]); s+=v; s2+=v*v; }
  r1[tid]=s; r2[tid]=s2; __syncthreads();
  for (int o=128;o;o>>=1){ if (tid<o){ r1[tid]+=r1[tid+o]; r2[tid]+=r2[tid+o]; } __syncthreads(); }
  float mean = r1[0]*(1.f/DD);
  float rstd = rsqrtf(r2[0]*(1.f/DD) - mean*mean + 1e-5f);
  for (int i=tid;i<DD;i+=256){
    float v=h[i]+gt[i]*(alpha[i]*f[i]);
    out[(size_t)row*DD + i] = (v-mean)*rstd;
  }
}

// ---------------- GAT edge kernels ----------------
__global__ __launch_bounds__(256) void edge_logits(
    const float* __restrict__ xl, const float* __restrict__ xr, const float* __restrict__ att,
    const int* __restrict__ ei, int NK, int E, float* __restrict__ logit, unsigned* __restrict__ mEnc)
{
  int w = (blockIdx.x * 256 + threadIdx.x) >> 5;
  if (w >= E * HH) return;
  int e = w >> 2, h = w & 3, lane = threadIdx.x & 31;
  int src, dst;
  if (e < NK){ src = ei[e]; dst = ei[NK + e]; } else { src = dst = e - NK; }
  float a = 0.f;
  for (int c = lane; c < CC; c += 32){
    float v = xl[(size_t)src*DD + h*CC + c] + xr[(size_t)dst*DD + h*CC + c];
    v = v > 0.f ? v : 0.2f * v;
    a += v * att[h*CC + c];
  }
  for (int o=16;o;o>>=1) a += __shfl_down(a, o, 32);
  if (!lane){
    logit[(size_t)e*HH + h] = a;
    atomicMax(&mEnc[dst*HH + h], fenc(a));
  }
}

__global__ __launch_bounds__(256) void edge_exp(
    const float* __restrict__ logit, const int* __restrict__ ei, int NK, int E,
    const unsigned* __restrict__ mEnc, float* __restrict__ ex, float* __restrict__ den)
{
  int i = blockIdx.x * 256 + threadIdx.x;
  if (i >= E * HH) return;
  int e = i >> 2, h = i & 3;
  int dst = (e < NK) ? ei[NK + e] : e - NK;
  float v = expf(logit[i] - fdec(mEnc[dst*HH + h]));
  ex[i] = v;
  atomicAdd(&den[dst*HH + h], v);
}

__global__ __launch_bounds__(256) void edge_scatter(
    const float* __restrict__ xl, const int* __restrict__ ei, int NK, int E,
    const float* __restrict__ ex, const float* __restrict__ den, float* __restrict__ outb)
{
  int w = (blockIdx.x * 256 + threadIdx.x) >> 5;
  if (w >= E * HH) return;
  int e = w >> 2, h = w & 3, lane = threadIdx.x & 31;
  int src, dst;
  if (e < NK){ src = ei[e]; dst = ei[NK + e]; } else { src = dst = e - NK; }
  float a = ex[(size_t)e*HH + h] / (den[dst*HH + h] + 1e-16f);
  for (int c = lane; c < CC; c += 32)
    atomicAdd(&outb[(size_t)dst*DD + h*CC + c], a * xl[(size_t)src*DD + h*CC + c]);
}

// ---------------- attention pooling ----------------
__global__ __launch_bounds__(256) void pool_score(
    const float* __restrict__ t, const float* __restrict__ W2, const float* __restrict__ b2,
    float* __restrict__ s, int Nn)
{
  int w = (blockIdx.x * 256 + threadIdx.x) >> 5;
  if (w >= Nn) return;
  int lane = threadIdx.x & 31;
  float a = 0.f;
  for (int c = lane; c < CC; c += 32) a += tanhf(t[(size_t)w*CC + c]) * W2[c];
  for (int o=16;o;o>>=1) a += __shfl_down(a, o, 32);
  if (!lane) s[w] = a + b2[0];
}

__global__ __launch_bounds__(256) void max_reduce(
    const float* __restrict__ s, int n, unsigned* __restrict__ out)
{
  __shared__ float r[256];
  float m = -3.402823466e38f;
  for (int i = blockIdx.x*256 + threadIdx.x; i < n; i += gridDim.x*256) m = fmaxf(m, s[i]);
  r[threadIdx.x] = m; __syncthreads();
  for (int o=128;o;o>>=1){ if (threadIdx.x<o) r[threadIdx.x]=fmaxf(r[threadIdx.x],r[threadIdx.x+o]); __syncthreads(); }
  if (!threadIdx.x) atomicMax(out, fenc(r[0]));
}

__global__ __launch_bounds__(256) void pool_gather(
    const float* __restrict__ x, const float* __restrict__ s, const unsigned* __restrict__ smax,
    int cnt, float* __restrict__ out)
{
  int g = blockIdx.x, tid = threadIdx.x;
  __shared__ float w[64]; __shared__ float den;
  float m = fdec(*smax);
  if (tid < cnt) w[tid] = expf(s[g*cnt + tid] - m);
  __syncthreads();
  if (!tid){ float d=0.f; for (int n=0;n<cnt;n++) d += w[n]; den = d + 1e-8f; }
  __syncthreads();
  for (int d = tid; d < DD; d += 256){
    float a = 0.f;
    for (int n = 0; n < cnt; n++) a += w[n] * x[((size_t)(g*cnt + n))*DD + d];
    out[(size_t)g*DD + d] = a / den;
  }
}

// ---------------- MHA core (L=112, per (b,h) block of 128 threads) ----------------
__global__ __launch_bounds__(128) void mha_attn(
    const float* __restrict__ qh, const float* __restrict__ kh, const float* __restrict__ vh,
    float* __restrict__ o, unsigned short* __restrict__ o_bf)
{
  int b = blockIdx.x >> 2, h = blockIdx.x & 3, tid = threadIdx.x;
  __shared__ float sc[LSEQ]; __shared__ float sm;
  if (tid < LSEQ){
    const float* q = qh + (size_t)b*DD + h*CC;
    const float* k = kh + ((size_t)b*LSEQ + tid)*DD + h*CC;
    float a = 0.f;
    for (int c=0;c<CC;c++) a += q[c]*k[c];
    sc[tid] = a * 0.08838834764831845f;  // 1/sqrt(128)
  }
  __syncthreads();
  if (!tid){
    float m = -3.402823466e38f;
    for (int l=0;l<LSEQ;l++) m = fmaxf(m, sc[l]);
    float s = 0.f;
    for (int l=0;l<LSEQ;l++){ float e = expf(sc[l]-m); sc[l]=e; s+=e; }
    sm = s;
  }
  __syncthreads();
  {
    int c = tid;
    float a = 0.f;
    for (int l=0;l<LSEQ;l++) a += sc[l] * vh[((size_t)b*LSEQ + l)*DD + h*CC + c];
    a /= sm;
    o[(size_t)b*DD + h*CC + c] = a;
    o_bf[(size_t)b*DD + h*CC + c] = f2bf(a);
  }
}

// ---------------- small utility kernels ----------------
__global__ void f32_to_bf16_k(const float* __restrict__ x, unsigned short* __restrict__ y, size_t n){
  for (size_t i = (size_t)blockIdx.x*256 + threadIdx.x; i < n; i += (size_t)gridDim.x*256)
    y[i] = f2bf(x[i]);
}
__global__ void fill_u32(unsigned* __restrict__ p, size_t n, unsigned v){
  for (size_t i = (size_t)blockIdx.x*256 + threadIdx.x; i < n; i += (size_t)gridDim.x*256)
    p[i] = v;
}
__global__ void add_dual(const float* __restrict__ a, const float* __restrict__ b,
                         float* __restrict__ y, unsigned short* __restrict__ ybf, size_t n){
  for (size_t i = (size_t)blockIdx.x*256 + threadIdx.x; i < n; i += (size_t)gridDim.x*256){
    float v = a[i] + b[i]; y[i] = v; ybf[i] = f2bf(v);
  }
}
__global__ void build_nodes(const float* __restrict__ hf, const float* __restrict__ ho,
                            unsigned short* __restrict__ nodes_bf){
  size_t i = (size_t)blockIdx.x*256 + threadIdx.x;
  if (i >= (size_t)BN*LSEQ*DD) return;
  int d = (int)(i & (DD-1));
  size_t r = i >> 9;
  int l = (int)(r % LSEQ), b = (int)(r / LSEQ);
  float v = (l < FN) ? hf[((size_t)b*FN + l)*DD + d] : ho[((size_t)b*ON + (l-FN))*DD + d];
  nodes_bf[i] = f2bf(v);
}
__global__ void build_z(const float* __restrict__ fs, const float* __restrict__ ff,
                        const float* __restrict__ fo, unsigned short* __restrict__ z){
  int i = blockIdx.x*256 + threadIdx.x;
  if (i >= BN*3*DD) return;
  int d = i % (3*DD), g = i / (3*DD);
  float v = (d < DD) ? fs[g*DD + d] : (d < 2*DD) ? ff[g*DD + d - DD] : fo[g*DD + d - 2*DD];
  z[i] = f2bf(v);
}
__global__ void head3(const float* __restrict__ X, const float* __restrict__ W,
                      const float* __restrict__ b, float* __restrict__ out){
  int g = blockIdx.x;
  if (threadIdx.x < 3){
    float a = 0.f;
    for (int d=0; d<DD; d++) a += X[(size_t)g*DD + d] * W[d*3 + threadIdx.x];
    out[g*3 + threadIdx.x] = a + b[threadIdx.x];
  }
}
__global__ __launch_bounds__(256) void mean_clf(
    const float* __restrict__ ho, const float* __restrict__ W, const float* __restrict__ b,
    float* __restrict__ out, int cnt){
  int g = blockIdx.x, tid = threadIdx.x;
  __shared__ float mn[DD];
  for (int d = tid; d < DD; d += 256){
    float a = 0.f;
    for (int n=0;n<cnt;n++) a += ho[((size_t)(g*cnt + n))*DD + d];
    mn[d] = a / (float)cnt;
  }
  __syncthreads();
  if (tid < 3){
    float a = 0.f;
    for (int d=0;d<DD;d++) a += mn[d] * W[d*3 + tid];
    out[g*3 + tid] = a + b[tid];
  }
}

// ---------------- host orchestration ----------------
extern "C" void kernel_launch(void* const* d_in, const int* in_sizes, int n_in,
                              void* d_out, int out_size, void* d_ws, size_t ws_size,
                              hipStream_t stream)
{
  (void)in_sizes; (void)n_in; (void)out_size; (void)ws_size;
  #define PF(i) ((const float*)d_in[(i)])
  #define PI(i) ((const int*)d_in[(i)])
  // input map (top-level: insertion order; params pytree: sorted keys)
  const float* face_x = PF(0);
  const float* obj_x  = PF(1);
  const float* scene_x= PF(2);
  // params: 3=Wg 4=alpha 5=bg 6,7=clf_c.{W,b} 8,9=clf_f 10,11=clf_s
  // 12..17 = clf_w.{W1,W2,b,b1,b2,g}
  // 18..25 = f_gat0.{Wl,Wr,att,bias,bl,br,nb,ng}  26..33 = f_gat1
  // 34..41 = mha.{Wk,Wo,Wq,Wv,bk,bo,bq,bv}
  // 42..49 = o_gat0  50..57 = o_gat1
  // 58..61 = pool_f.{W1,W2,b1,b2} 62..65 = pool_fbr 66..69 = pool_o
  // 70..75 = red_f.{W,b,b1,b2,g1,g2} 76..81 = red_o 82..87 = red_s
  const int* face_ei   = PI(88);
  const int* obj_ei    = PI(89);
  const int* face_bat  = PI(90);
  float* out = (float*)d_out;

  // workspace allocator
  size_t off = 0;
  auto alloc = [&](size_t bytes)->void*{
    void* p = (void*)((char*)d_ws + off);
    off += (bytes + 255) & ~(size_t)255;
    return p;
  };
  auto U16 = [&](size_t n){ return (unsigned short*)alloc(n*2); };
  auto F32 = [&](size_t n){ return (float*)alloc(n*4); };
  auto U32n= [&](size_t n){ return (unsigned*)alloc(n*4); };

  unsigned short* wred_f = U16((size_t)4096*DD);
  unsigned short* wred_o = U16((size_t)2048*DD);
  unsigned short* wred_s = U16((size_t)1024*DD);
  unsigned short* wgat[8];
  for (int i=0;i<8;i++) wgat[i] = U16((size_t)DD*DD);
  unsigned short* wg_bf  = U16((size_t)DD*DD);
  unsigned short* wmk = U16((size_t)DD*DD);
  unsigned short* wmo = U16((size_t)DD*DD);
  unsigned short* wmq = U16((size_t)DD*DD);
  unsigned short* wmv = U16((size_t)DD*DD);
  unsigned short* wp1_f   = U16((size_t)DD*CC);
  unsigned short* wp1_fbr = U16((size_t)DD*CC);
  unsigned short* wp1_o   = U16((size_t)DD*CC);
  unsigned short* wz1     = U16((size_t)3*DD*DD);

  unsigned short* xbig = U16((size_t)NF*4096);  // reused: LN'ed inputs, nodes_bf, z_bf
  float* pf = F32((size_t)NF*DD); unsigned short* pf_bf = U16((size_t)NF*DD);
  float* po = F32((size_t)NO*DD); unsigned short* po_bf = U16((size_t)NO*DD);
  float* ps = F32((size_t)BN*DD); unsigned short* ps_bf = U16((size_t)BN*DD);
  float* hf = F32((size_t)NF*DD); unsigned short* hf_bf = U16((size_t)NF*DD);
  float* ho = F32((size_t)NO*DD); unsigned short* ho_bf = U16((size_t)NO*DD);
  float* xl = F32((size_t)NO*DD);
  float* xr = F32((size_t)NO*DD);
  float* gout = F32((size_t)NO*DD);
  float* logit = F32((size_t)EO*HH);
  float* exb   = F32((size_t)EO*HH);
  unsigned* mEnc = U32n((size_t)NO*HH);
  float* den     = F32((size_t)NO*HH);
  float* kh = F32((size_t)BN*LSEQ*DD);
  float* vh = F32((size_t)BN*LSEQ*DD);
  float* qh = F32((size_t)BN*DD);
  float* oatt = F32((size_t)BN*DD); unsigned short* oatt_bf = U16((size_t)BN*DD);
  float* attn = F32((size_t)BN*DD);
  float* fs   = F32((size_t)BN*DD);
  float* gate = F32((size_t)NF*DD);
  float* hf2  = F32((size_t)NF*DD);
  float* xpf = F32((size_t)NF*DD); unsigned short* xpf_bf = U16((size_t)NF*DD);
  float* xpo = F32((size_t)NO*DD); unsigned short* xpo_bf = U16((size_t)NO*DD);
  float* tpool = F32((size_t)NO*CC);
  float* svec  = F32((size_t)NO);
  unsigned* smax = U32n(64);
  float* pooled = F32((size_t)BN*DD);
  float* featf  = F32((size_t)BN*DD);
  float* feato  = F32((size_t)BN*DD);
  float* t1     = F32((size_t)BN*DD);

  auto tobf = [&](const float* s, unsigned short* d, size_t n){
    unsigned gb = (unsigned)(((n + 255)/256) < 16384 ? ((n + 255)/256) : 16384);
    f32_to_bf16_k<<<gb, 256, 0, stream>>>(s, d, n);
  };
  auto zero = [&](void* p, size_t n32){
    unsigned gb = (unsigned)(((n32 + 255)/256) < 16384 ? ((n32 + 255)/256) : 16384);
    fill_u32<<<gb, 256, 0, stream>>>((unsigned*)p, n32, 0u);
  };
  auto gemm = [&](const unsigned short* A, const unsigned short* B, const float* bias,
                  float* C, int M, int N, int K, int act){
    wmma_gemm<<<dim3((unsigned)(N/128), (unsigned)(M/128)), 256, 0, stream>>>(A, B, bias, C, M, N, K, act);
  };

  // ---- convert weights to bf16 ----
  tobf(PF(70), wred_f, (size_t)4096*DD);
  tobf(PF(76), wred_o, (size_t)2048*DD);
  tobf(PF(82), wred_s, (size_t)1024*DD);
  tobf(PF(18), wgat[0], (size_t)DD*DD); tobf(PF(19), wgat[1], (size_t)DD*DD);
  tobf(PF(26), wgat[2], (size_t)DD*DD); tobf(PF(27), wgat[3], (size_t)DD*DD);
  tobf(PF(42), wgat[4], (size_t)DD*DD); tobf(PF(43), wgat[5], (size_t)DD*DD);
  tobf(PF(50), wgat[6], (size_t)DD*DD); tobf(PF(51), wgat[7], (size_t)DD*DD);
  tobf(PF(3),  wg_bf,  (size_t)DD*DD);
  tobf(PF(34), wmk, (size_t)DD*DD); tobf(PF(35), wmo, (size_t)DD*DD);
  tobf(PF(36), wmq, (size_t)DD*DD); tobf(PF(37), wmv, (size_t)DD*DD);
  tobf(PF(58), wp1_f,   (size_t)DD*CC);
  tobf(PF(62), wp1_fbr, (size_t)DD*CC);
  tobf(PF(66), wp1_o,   (size_t)DD*CC);
  tobf(PF(12), wz1, (size_t)3*DD*DD);

  // ---- _red reductions ----
  ln_rows_bf16<<<NF, 256, 0, stream>>>(face_x, PF(74), PF(72), xbig, 4096);
  gemm(xbig, wred_f, PF(71), pf, NF, DD, 4096, 0);
  ln_act_dual<<<NF, 256, 0, stream>>>(pf, PF(75), PF(73), pf, pf_bf, 1);

  ln_rows_bf16<<<NO, 256, 0, stream>>>(obj_x, PF(80), PF(78), xbig, 2048);
  gemm(xbig, wred_o, PF(77), po, NO, DD, 2048, 0);
  ln_act_dual<<<NO, 256, 0, stream>>>(po, PF(81), PF(79), po, po_bf, 1);

  ln_rows_bf16<<<BN, 256, 0, stream>>>(scene_x, PF(86), PF(84), xbig, 1024);
  gemm(xbig, wred_s, PF(83), ps, BN, DD, 1024, 0);
  ln_act_dual<<<BN, 256, 0, stream>>>(ps, PF(87), PF(85), ps, ps_bf, 1);

  // ---- GAT stacks ----
  auto gat_layer = [&](const float* hin, const unsigned short* hin_bf,
                       float* hout, unsigned short* hout_bf,
                       int N, int NK, int E, const int* ei,
                       const unsigned short* Wl, const unsigned short* Wr, int pbase){
    gemm(hin_bf, Wl, PF(pbase+4), xl, N, DD, DD, 0);
    gemm(hin_bf, Wr, PF(pbase+5), xr, N, DD, DD, 0);
    zero(mEnc, (size_t)N*HH);
    zero(den,  (size_t)N*HH);
    zero(gout, (size_t)N*DD);
    int warps = E * HH;
    edge_logits<<<(unsigned)((warps + 7)/8), 256, 0, stream>>>(xl, xr, PF(pbase+2), ei, NK, E, logit, mEnc);
    edge_exp<<<(unsigned)((warps + 255)/256), 256, 0, stream>>>(logit, ei, NK, E, mEnc, exb, den);
    edge_scatter<<<(unsigned)((warps + 7)/8), 256, 0, stream>>>(xl, ei, NK, E, exb, den, gout);
    gat_post<<<N, 256, 0, stream>>>(hin, gout, PF(pbase+3), PF(pbase+7), PF(pbase+6), hout, hout_bf);
  };
  gat_layer(pf, pf_bf, hf, hf_bf, NF, NKF, EF, face_ei, wgat[0], wgat[1], 18);
  gat_layer(hf, hf_bf, hf, hf_bf, NF, NKF, EF, face_ei, wgat[2], wgat[3], 26);
  gat_layer(po, po_bf, ho, ho_bf, NO, NKO, EO, obj_ei,  wgat[4], wgat[5], 42);
  gat_layer(ho, ho_bf, ho, ho_bf, NO, NKO, EO, obj_ei,  wgat[6], wgat[7], 50);

  // ---- attnpool helper ----
  auto attnpool = [&](const unsigned short* xbf, const float* xf, int N, int cnt,
                      const unsigned short* W1bf, int pbase, float* outp){
    gemm(xbf, W1bf, PF(pbase+2), tpool, N, CC, DD, 0);
    pool_score<<<(unsigned)(N/8), 256, 0, stream>>>(tpool, PF(pbase+1), PF(pbase+3), svec, N);
    zero(smax, 1);
    max_reduce<<<256, 256, 0, stream>>>(svec, N, smax);
    pool_gather<<<BN, 256, 0, stream>>>(xf, svec, smax, cnt, outp);
  };

  // ---- out_f, out_c, out_s ----
  attnpool(hf_bf, hf, NF, FN, wp1_fbr, 62, pooled);
  head3<<<BN, 64, 0, stream>>>(pooled, PF(8), PF(9), out + 0);            // out_f
  mean_clf<<<BN, 256, 0, stream>>>(ho, PF(6), PF(7), out + BN*3, ON);     // out_c
  head3<<<BN, 64, 0, stream>>>(ps, PF(10), PF(11), out + 2*BN*3);        // out_s

  // ---- MHA over [face_nodes; obj_nodes] ----
  build_nodes<<<(unsigned)(((size_t)BN*LSEQ*DD + 255)/256), 256, 0, stream>>>(hf, ho, xbig);
  gemm(ps_bf, wmq, PF(40), qh, BN, DD, DD, 0);
  gemm(xbig, wmk, PF(38), kh, BN*LSEQ, DD, DD, 0);
  gemm(xbig, wmv, PF(41), vh, BN*LSEQ, DD, DD, 0);
  mha_attn<<<BN*HH, 128, 0, stream>>>(qh, kh, vh, oatt, oatt_bf);
  gemm(oatt_bf, wmo, PF(39), attn, BN, DD, DD, 0);
  add_lnplain<<<BN, 256, 0, stream>>>(ps, attn, fs);

  // ---- gated face fusion ----
  gemm(hf_bf, wg_bf, PF(5), gate, NF, DD, DD, 2);  // sigmoid
  face_fuse<<<NF, 256, 0, stream>>>(hf, gate, PF(4), fs, face_bat, hf2);
  {
    unsigned gb = (unsigned)(((size_t)NF*DD + 255)/256);
    add_dual<<<gb, 256, 0, stream>>>(hf2, pf, xpf, xpf_bf, (size_t)NF*DD);
    gb = (unsigned)(((size_t)NO*DD + 255)/256);
    add_dual<<<gb, 256, 0, stream>>>(ho, po, xpo, xpo_bf, (size_t)NO*DD);
  }
  attnpool(xpf_bf, xpf, NF, FN, wp1_f, 58, featf);
  attnpool(xpo_bf, xpo, NO, ON, wp1_o, 66, feato);

  // ---- out_w head ----
  build_z<<<(unsigned)((BN*3*DD + 255)/256), 256, 0, stream>>>(fs, featf, feato, xbig);
  gemm(xbig, wz1, PF(15), t1, BN, DD, 3*DD, 0);
  ln_act_dual<<<BN, 256, 0, stream>>>(t1, PF(17), PF(14), t1, nullptr, 1);
  head3<<<BN, 64, 0, stream>>>(t1, PF(13), PF(16), out + 3*BN*3);        // out_w

  #undef PF
  #undef PI
}
